// DeepSeekMoE_47236050321676
// MI455X (gfx1250) — compile-verified, matrix operand order checked by reading, not run on hardware
//
#include <hip/hip_runtime.h>
#include <math.h>

// ---------------- problem constants ----------------
constexpr int TOKENS = 2048;   // B*S
constexpr int DDIM   = 1024;
constexpr int NEXP   = 8;
constexpr int TOPK   = 2;
constexpr int HDIM   = 4096;
constexpr int RMAX   = TOKENS * TOPK + NEXP * 64;   // 4608: padded row capacity

// ---------------- workspace layout (bytes) ----------------
constexpr size_t OFF_COUNTS = 0;
constexpr size_t OFF_CURS   = 256;
constexpr size_t OFF_OFFS   = 512;                                   // NEXP+1 ints
constexpr size_t OFF_TOPI   = 1024;
constexpr size_t OFF_TOPW   = OFF_TOPI + (size_t)TOKENS * 2 * 4;
constexpr size_t OFF_RMAP   = OFF_TOPW + (size_t)TOKENS * 2 * 4;
constexpr size_t OFF_TOK    = OFF_RMAP + (size_t)TOKENS * 2 * 4;
constexpr size_t OFF_H      = (OFF_TOK + (size_t)RMAX * 4 + 255) & ~(size_t)255;
constexpr size_t OFF_EO     = OFF_H  + (size_t)RMAX * HDIM * 2;      // bf16 h
constexpr size_t OFF_HS     = OFF_EO + (size_t)RMAX * DDIM * 4;      // f32 eo
// total ~ 73.5 MB

typedef __attribute__((ext_vector_type(16))) __bf16 bf16x16;
typedef __attribute__((ext_vector_type(8)))  float  f32x8;

__device__ __forceinline__ float gelu_exact(float v) {
    return 0.5f * v * (1.0f + erff(v * 0.70710678118654752f));
}

// ---------------- routing kernels ----------------
__global__ void moe_init(int* counts, int* tok_of_row) {
    int i = blockIdx.x * blockDim.x + threadIdx.x;
    if (i < NEXP) counts[i] = 0;
    if (i < RMAX) tok_of_row[i] = -1;
}

__global__ void moe_gate(const float* __restrict__ x, const float* __restrict__ Wg,
                         const float* __restrict__ bg, const float* __restrict__ rbias,
                         int* __restrict__ top_idx, float* __restrict__ top_w,
                         int* __restrict__ counts) {
    int t = blockIdx.x;
    int wave = threadIdx.x >> 5, lane = threadIdx.x & 31;
    const float* xr = x + (size_t)t * DDIM;
    const float* wr = Wg + (size_t)wave * DDIM;       // 8 waves == 8 experts
    float s = 0.f;
    for (int i = lane; i < DDIM; i += 32) s += xr[i] * wr[i];
    for (int o = 16; o; o >>= 1) s += __shfl_xor(s, o, 32);
    __shared__ float sc[NEXP];
    if (lane == 0) sc[wave] = 1.0f / (1.0f + expf(-(s + bg[wave] + rbias[wave])));
    __syncthreads();
    if (threadIdx.x == 0) {
        int i0 = 0, i1 = -1; float s0 = -1e30f, s1 = -1e30f;
        for (int e = 0; e < NEXP; ++e) {
            float v = sc[e];
            if (v > s0)      { s1 = s0; i1 = i0; s0 = v; i0 = e; }
            else if (v > s1) { s1 = v;  i1 = e; }
        }
        top_idx[t * 2] = i0; top_idx[t * 2 + 1] = i1;
        top_w[t * 2]   = s0; top_w[t * 2 + 1]   = s1;
        atomicAdd(&counts[i0], 1);
        atomicAdd(&counts[i1], 1);
    }
}

__global__ void moe_scan(const int* __restrict__ counts, int* __restrict__ offs,
                         int* __restrict__ cursors) {
    if (threadIdx.x == 0) {
        int off = 0;
        for (int e = 0; e < NEXP; ++e) {
            offs[e] = off; cursors[e] = off;
            off += (counts[e] + 63) & ~63;           // pad each expert to 64-row tiles
        }
        offs[NEXP] = off;
    }
}

__global__ void moe_build(const int* __restrict__ top_idx, int* __restrict__ cursors,
                          int* __restrict__ rowmap, int* __restrict__ tok_of_row) {
    int t = blockIdx.x * blockDim.x + threadIdx.x;
    if (t >= TOKENS) return;
    for (int k = 0; k < TOPK; ++k) {
        int e = top_idx[t * 2 + k];
        int row = atomicAdd(&cursors[e], 1);
        rowmap[t * 2 + k] = row;
        tok_of_row[row] = t;
    }
}

// 16-bit fragment K-interleave (CDNA5 A/B layout)
__device__ __forceinline__ int frag_kk(int j, int half) {
    return (j & 7) + ((j >> 3) << 4) + half * 8;
}

#define WMMA_BF16(A, B, C) \
    __builtin_amdgcn_wmma_f32_16x16x32_bf16(false, (A), false, (B), (short)0, (C), false, false)

// ---------------- GEMM 1: fp32 (gathered) A, fused GELU, bf16 out ----------------
// block tile 64x128, wave tile 32x32, BK=32, double-buffered LDS
__global__ __launch_bounds__(256)
void moe_gemm_fp32A(const float* __restrict__ X, const float* __restrict__ W,
                    const float* __restrict__ bias, __bf16* __restrict__ Hout,
                    const int* __restrict__ tok_of_row, const int* __restrict__ offs,
                    int Kdim, int Ndim, int gather) {
    __shared__ __bf16 As[2][64][40];     // [m][k] ; 80B row stride (16B-aligned)
    __shared__ __bf16 Bs[2][128][40];    // [n][k]
    int row_base = blockIdx.x * 64;
    if (gather && row_base >= offs[NEXP]) return;    // uniform early-out
    int e = 0;
    if (gather) { while (e < NEXP - 1 && row_base >= offs[e + 1]) ++e; }
    const float* We = W + (size_t)e * Ndim * Kdim;   // torch [out,in] layout
    const float* be = bias + (size_t)e * Ndim;
    int nbase = blockIdx.y * 128;

    int tid = threadIdx.x, lane = tid & 31, wave = tid >> 5;
    int half = lane >> 4, l16 = lane & 15;
    int wm = (wave & 1) * 32;            // wave's 32-row strip
    int wn = (wave >> 1) * 32;           // wave's 32-col strip

    // staging maps: A 64x32 (8 f32/thr), B 128x32 (16 f32/thr)
    int arow = tid >> 2, acol = (tid & 3) * 8;
    int brow = tid >> 1, bcol = (tid & 1) * 16;

    int tokr = gather ? tok_of_row[row_base + arow] : (row_base + arow);
    bool avalid = tokr >= 0;
    const float* xrow = X + (size_t)(avalid ? tokr : 0) * Kdim + acol;   // clamped, branch-free
    const float* wrow = We + (size_t)(nbase + brow) * Kdim + bcol;

    float areg[8], breg[16];
    {   // prologue: global -> regs for tile 0
        float4 a0 = *(const float4*)(xrow + 0), a1 = *(const float4*)(xrow + 4);
        areg[0]=a0.x; areg[1]=a0.y; areg[2]=a0.z; areg[3]=a0.w;
        areg[4]=a1.x; areg[5]=a1.y; areg[6]=a1.z; areg[7]=a1.w;
        #pragma unroll
        for (int q = 0; q < 4; ++q) {
            float4 b = *(const float4*)(wrow + q * 4);
            breg[q*4+0]=b.x; breg[q*4+1]=b.y; breg[q*4+2]=b.z; breg[q*4+3]=b.w;
        }
    }

    f32x8 acc00 = {}, acc01 = {}, acc10 = {}, acc11 = {};
    int nk = Kdim >> 5;
    for (int t = 0; t < nk; ++t) {
        int buf = t & 1;
        // regs -> LDS[buf] (convert fp32 -> bf16; zero-fill gathered-invalid rows)
        #pragma unroll
        for (int i = 0; i < 8; ++i) As[buf][arow][acol + i] = avalid ? (__bf16)areg[i] : (__bf16)0.0f;
        #pragma unroll
        for (int i = 0; i < 16; ++i) Bs[buf][brow][bcol + i] = (__bf16)breg[i];
        __syncthreads();
        // issue next tile's global loads (overlaps WMMA below)
        if (t + 1 < nk) {
            const float* xn = xrow + (t + 1) * 32;
            const float* wn_ = wrow + (t + 1) * 32;
            float4 a0 = *(const float4*)(xn + 0), a1 = *(const float4*)(xn + 4);
            areg[0]=a0.x; areg[1]=a0.y; areg[2]=a0.z; areg[3]=a0.w;
            areg[4]=a1.x; areg[5]=a1.y; areg[6]=a1.z; areg[7]=a1.w;
            #pragma unroll
            for (int q = 0; q < 4; ++q) {
                float4 b = *(const float4*)(wn_ + q * 4);
                breg[q*4+0]=b.x; breg[q*4+1]=b.y; breg[q*4+2]=b.z; breg[q*4+3]=b.w;
            }
        }
        if (t + 2 < nk) __builtin_prefetch(wrow + (t + 2) * 32, 0, 1);
        // fragments + 4 WMMA (A fragments reused across two B columns)
        bf16x16 a0, a1, bb;
        #pragma unroll
        for (int j = 0; j < 16; ++j) {
            int kk = frag_kk(j, half);
            a0[j] = As[buf][wm + l16][kk];
            a1[j] = As[buf][wm + 16 + l16][kk];
        }
        #pragma unroll
        for (int j = 0; j < 16; ++j) bb[j] = Bs[buf][wn + l16][frag_kk(j, half)];
        acc00 = WMMA_BF16(a0, bb, acc00);
        acc10 = WMMA_BF16(a1, bb, acc10);
        #pragma unroll
        for (int j = 0; j < 16; ++j) bb[j] = Bs[buf][wn + 16 + l16][frag_kk(j, half)];
        acc01 = WMMA_BF16(a0, bb, acc01);
        acc11 = WMMA_BF16(a1, bb, acc11);
    }

    // epilogue: +bias, GELU, bf16 store
    #pragma unroll
    for (int i = 0; i < 8; ++i) {
        int m0 = row_base + wm + i + 8 * half;        // acc00/acc01 rows
        int m1 = m0 + 16;                              // acc10/acc11 rows
        int n0 = nbase + wn + l16;
        int n1 = n0 + 16;
        Hout[(size_t)m0 * Ndim + n0] = (__bf16)gelu_exact(acc00[i] + be[n0]);
        Hout[(size_t)m0 * Ndim + n1] = (__bf16)gelu_exact(acc01[i] + be[n1]);
        Hout[(size_t)m1 * Ndim + n0] = (__bf16)gelu_exact(acc10[i] + be[n0]);
        Hout[(size_t)m1 * Ndim + n1] = (__bf16)gelu_exact(acc11[i] + be[n1]);
    }
}

// ---------------- GEMM 2: bf16 A, fp32 out + bias ----------------
__global__ __launch_bounds__(256)
void moe_gemm_bf16A(const __bf16* __restrict__ A, const float* __restrict__ W,
                    const float* __restrict__ bias, float* __restrict__ Out,
                    const int* __restrict__ offs, int Kdim, int Ndim, int gather) {
    __shared__ __bf16 As[2][64][40];
    __shared__ __bf16 Bs[2][128][40];
    int row_base = blockIdx.x * 64;
    if (gather && row_base >= offs[NEXP]) return;
    int e = 0;
    if (gather) { while (e < NEXP - 1 && row_base >= offs[e + 1]) ++e; }
    const float* We = W + (size_t)e * Ndim * Kdim;
    const float* be = bias + (size_t)e * Ndim;
    int nbase = blockIdx.y * 128;

    int tid = threadIdx.x, lane = tid & 31, wave = tid >> 5;
    int half = lane >> 4, l16 = lane & 15;
    int wm = (wave & 1) * 32;
    int wn = (wave >> 1) * 32;

    int arow = tid >> 2, acol = (tid & 3) * 8;        // A: 8 bf16/thr (16B)
    int brow = tid >> 1, bcol = (tid & 1) * 16;       // B: 16 f32/thr

    const __bf16* arp = A + (size_t)(row_base + arow) * Kdim + acol;
    const float*  wrow = We + (size_t)(nbase + brow) * Kdim + bcol;

    typedef __attribute__((ext_vector_type(8))) __bf16 bf16x8;
    bf16x8 areg;
    float breg[16];
    {
        areg = *(const bf16x8*)arp;
        #pragma unroll
        for (int q = 0; q < 4; ++q) {
            float4 b = *(const float4*)(wrow + q * 4);
            breg[q*4+0]=b.x; breg[q*4+1]=b.y; breg[q*4+2]=b.z; breg[q*4+3]=b.w;
        }
    }

    f32x8 acc00 = {}, acc01 = {}, acc10 = {}, acc11 = {};
    int nk = Kdim >> 5;
    for (int t = 0; t < nk; ++t) {
        int buf = t & 1;
        *(bf16x8*)&As[buf][arow][acol] = areg;
        #pragma unroll
        for (int i = 0; i < 16; ++i) Bs[buf][brow][bcol + i] = (__bf16)breg[i];
        __syncthreads();
        if (t + 1 < nk) {
            areg = *(const bf16x8*)(arp + (t + 1) * 32);
            const float* wn_ = wrow + (t + 1) * 32;
            #pragma unroll
            for (int q = 0; q < 4; ++q) {
                float4 b = *(const float4*)(wn_ + q * 4);
                breg[q*4+0]=b.x; breg[q*4+1]=b.y; breg[q*4+2]=b.z; breg[q*4+3]=b.w;
            }
        }
        if (t + 2 < nk) __builtin_prefetch(wrow + (t + 2) * 32, 0, 1);
        bf16x16 a0, a1, bb;
        #pragma unroll
        for (int j = 0; j < 16; ++j) {
            int kk = frag_kk(j, half);
            a0[j] = As[buf][wm + l16][kk];
            a1[j] = As[buf][wm + 16 + l16][kk];
        }
        #pragma unroll
        for (int j = 0; j < 16; ++j) bb[j] = Bs[buf][wn + l16][frag_kk(j, half)];
        acc00 = WMMA_BF16(a0, bb, acc00);
        acc10 = WMMA_BF16(a1, bb, acc10);
        #pragma unroll
        for (int j = 0; j < 16; ++j) bb[j] = Bs[buf][wn + 16 + l16][frag_kk(j, half)];
        acc01 = WMMA_BF16(a0, bb, acc01);
        acc11 = WMMA_BF16(a1, bb, acc11);
    }

    #pragma unroll
    for (int i = 0; i < 8; ++i) {
        int m0 = row_base + wm + i + 8 * half;
        int m1 = m0 + 16;
        int n0 = nbase + wn + l16;
        int n1 = n0 + 16;
        Out[(size_t)m0 * Ndim + n0] = acc00[i] + be[n0];
        Out[(size_t)m0 * Ndim + n1] = acc01[i] + be[n1];
        Out[(size_t)m1 * Ndim + n0] = acc10[i] + be[n0];
        Out[(size_t)m1 * Ndim + n1] = acc11[i] + be[n1];
    }
}

// ---------------- deterministic combine ----------------
__global__ void moe_combine(float* __restrict__ out, const float* __restrict__ eo,
                            const int* __restrict__ rowmap, const float* __restrict__ top_w) {
    int idx = blockIdx.x * blockDim.x + threadIdx.x;   // < TOKENS*DDIM
    int t = idx / DDIM, d = idx - t * DDIM;
    float v = out[idx];                                 // shared-expert result
    v += top_w[t * 2]     * eo[(size_t)rowmap[t * 2]     * DDIM + d];
    v += top_w[t * 2 + 1] * eo[(size_t)rowmap[t * 2 + 1] * DDIM + d];
    out[idx] = v;
}

// ---------------- launcher ----------------
extern "C" void kernel_launch(void* const* d_in, const int* in_sizes, int n_in,
                              void* d_out, int out_size, void* d_ws, size_t ws_size,
                              hipStream_t stream) {
    (void)in_sizes; (void)n_in; (void)out_size; (void)ws_size;
    const float* x   = (const float*)d_in[0];
    const float* Wg  = (const float*)d_in[1];
    const float* bg  = (const float*)d_in[2];
    const float* rb  = (const float*)d_in[3];
    const float* W1  = (const float*)d_in[4];
    const float* b1  = (const float*)d_in[5];
    const float* W2  = (const float*)d_in[6];
    const float* b2  = (const float*)d_in[7];
    const float* Ws1 = (const float*)d_in[8];
    const float* bs1 = (const float*)d_in[9];
    const float* Ws2 = (const float*)d_in[10];
    const float* bs2 = (const float*)d_in[11];
    float* out = (float*)d_out;

    char* ws = (char*)d_ws;
    int*    counts  = (int*)(ws + OFF_COUNTS);
    int*    cursors = (int*)(ws + OFF_CURS);
    int*    offs    = (int*)(ws + OFF_OFFS);
    int*    top_idx = (int*)(ws + OFF_TOPI);
    float*  top_w   = (float*)(ws + OFF_TOPW);
    int*    rowmap  = (int*)(ws + OFF_RMAP);
    int*    tok_row = (int*)(ws + OFF_TOK);
    __bf16* h_rt    = (__bf16*)(ws + OFF_H);
    float*  eo      = (float*)(ws + OFF_EO);
    __bf16* hS      = (__bf16*)(ws + OFF_HS);

    moe_init <<<RMAX / 256, 256, 0, stream>>>(counts, tok_row);
    moe_gate <<<TOKENS, 256, 0, stream>>>(x, Wg, bg, rb, top_idx, top_w, counts);
    moe_scan <<<1, 32, 0, stream>>>(counts, offs, cursors);
    moe_build<<<TOKENS / 256, 256, 0, stream>>>(top_idx, cursors, rowmap, tok_row);

    // routed experts: h = gelu(x @ W1^T + b1) ; eo = h @ W2^T + b2
    dim3 g1(RMAX / 64, HDIM / 128);
    moe_gemm_fp32A<<<g1, 256, 0, stream>>>(x, W1, b1, h_rt, tok_row, offs, DDIM, HDIM, 1);
    dim3 g2(RMAX / 64, DDIM / 128);
    moe_gemm_bf16A<<<g2, 256, 0, stream>>>(h_rt, W2, b2, eo, offs, HDIM, DDIM, 1);

    // shared expert: writes d_out directly
    dim3 g3(TOKENS / 64, HDIM / 128);
    moe_gemm_fp32A<<<g3, 256, 0, stream>>>(x, Ws1, bs1, hS, nullptr, offs, DDIM, HDIM, 0);
    dim3 g4(TOKENS / 64, DDIM / 128);
    moe_gemm_bf16A<<<g4, 256, 0, stream>>>(hS, Ws2, bs2, out, offs, HDIM, DDIM, 0);

    moe_combine<<<(TOKENS * DDIM) / 256, 256, 0, stream>>>(out, eo, rowmap, top_w);
}
